// SNAPSGNN_90941637525591
// MI455X (gfx1250) — matrix operand order repaired and checked by reading, not compiled, not run on hardware
//
#include <hip/hip_runtime.h>
#include <hip/hip_bf16.h>

typedef __attribute__((ext_vector_type(16))) _Float16 v16h;
typedef __attribute__((ext_vector_type(8)))  float    v8f;

#define EPS_BN 1e-5f

// ---------------------------------------------------------------------------
// Degree / normalization
// ---------------------------------------------------------------------------
__global__ void k_deg_init(float* __restrict__ deg, int n) {
    int i = blockIdx.x * blockDim.x + threadIdx.x;
    if (i < n) deg[i] = 1.0f;                       // self-loop contribution
}

__global__ void k_deg_edges(const int* __restrict__ dst, float* __restrict__ deg,
                            long long e_cnt) {
    long long e = (long long)blockIdx.x * blockDim.x + threadIdx.x;
    if (e < e_cnt) atomicAdd(&deg[dst[e]], 1.0f);
}

__global__ void k_dis(float* __restrict__ deg, int n) {
    int i = blockIdx.x * blockDim.x + threadIdx.x;
    if (i < n) {
        float d = deg[i];
        deg[i] = rsqrtf(fmaxf(d, 1e-12f));          // in place: deg -> dis
    }
}

// ---------------------------------------------------------------------------
// Pack W (f32 row-major [128][FOUT]) into f16 WMMA B-fragment order:
//   PW[((kb*NCT + ct)*32 + lane)*16 + m] = W[kb*32 + 16*(m>>3) + 8*(lane>>4) + (m&7)][ct*16 + (lane&15)]
// so each wave's per-K-step B fragment is one contiguous 32B chunk per lane.
// ---------------------------------------------------------------------------
template<int FOUT>
__global__ void k_packW(const float* __restrict__ W, _Float16* __restrict__ PW) {
    constexpr int NCT = (FOUT + 15) / 16;
    const int total = 4 * NCT * 32 * 16;
    int t = blockIdx.x * blockDim.x + threadIdx.x;
    if (t >= total) return;
    int m    = t & 15;
    int lane = (t >> 4) & 31;
    int q    = t >> 9;              // kb*NCT + ct
    int ct   = q % NCT;
    int kb   = q / NCT;
    int krow = kb * 32 + ((m >> 3) << 4) + ((lane >> 4) << 3) + (m & 7);
    int col  = ct * 16 + (lane & 15);
    float v  = (col < FOUT) ? W[(size_t)krow * FOUT + col] : 0.0f;
    PW[t] = (_Float16)v;
}

// ---------------------------------------------------------------------------
// Dense GEMM  H = X @ W  via V_WMMA_F32_16X16X32_F16.
// Block = 256 threads = 8 waves, one 16-row tile per block (grid = nrows/16,
// nrows is a multiple of 16 so no row guards anywhere). The block first
// cooperatively converts the 16x128 f32 A tile into LDS f16 (rows padded to
// 136 halves = 272B: 16B-aligned rows, staggered banks). Wave w then owns
// column tile w; per K-step it reads A as two ds_load_b128 and B as one
// contiguous 32B global load from the pre-packed fragment buffer.
// ---------------------------------------------------------------------------
template<int FOUT>
__global__ void k_gemm(const float* __restrict__ X, const _Float16* __restrict__ PW,
                       float* __restrict__ H) {
    constexpr int FIN = 128;
    constexpr int NCT = (FOUT + 15) / 16;
    __shared__ _Float16 As[16][136];                // 272B row stride

    const int tid  = threadIdx.x;
    const int row0 = blockIdx.x * 16;

    // cooperative A-tile stage: 256 threads x 8 halves
    {
        int r  = tid >> 4;                          // 0..15
        int kk = (tid & 15) << 3;                   // 0,8,...,120
        const float* xp = X + (size_t)(row0 + r) * FIN + kk;
        #pragma unroll
        for (int j = 0; j < 8; ++j) As[r][kk + j] = (_Float16)xp[j];
    }
    __syncthreads();

    const int wave = tid >> 5;
    if (wave >= NCT) return;                        // uniform per wave
    const int lane = tid & 31;
    const int g    = lane >> 4;
    const int l16  = lane & 15;

    v8f acc = {};
    #pragma unroll
    for (int kb = 0; kb < FIN / 32; ++kb) {
        const int k = kb * 32;
        v16h a, b;
        const _Float16* a0 = &As[l16][k + 8 * g];        // 16B aligned
        const _Float16* a1 = &As[l16][k + 16 + 8 * g];   // 16B aligned
        #pragma unroll
        for (int j = 0; j < 8; ++j) { a[j] = a0[j]; a[8 + j] = a1[j]; }
        b = *(const v16h*)(PW + ((size_t)(kb * NCT + wave) * 32 + lane) * 16);
        acc = __builtin_amdgcn_wmma_f32_16x16x32_f16(
                  false, a, false, b, (short)0, acc, false, false);
    }

    // C/D layout: VGPR v, lanes 0-15 -> M=v, lanes 16-31 -> M=v+8; N=l16
    const int col = wave * 16 + l16;
    float* hp = H + (size_t)(row0 + 8 * g) * FOUT + col;
    if (col < FOUT) {                               // constexpr-true for FOUT=128
        #pragma unroll
        for (int v = 0; v < 8; ++v)
            hp[(size_t)v * FOUT] = acc[v];
    }
}

// ---------------------------------------------------------------------------
// agg[i][c] = H[i][c] * dis[i]^2   (self-loop message == aggregation init,
// avoids a separate zero-fill pass over the 51MB buffer)
// ---------------------------------------------------------------------------
template<int F>
__global__ void k_selfloop(const float* __restrict__ H, const float* __restrict__ dis,
                           float* __restrict__ agg, int n) {
    long long tid = (long long)blockIdx.x * blockDim.x + threadIdx.x;
    if (tid >= (long long)n * F) return;
    int i = (int)(tid / F);
    float d = dis[i];
    agg[tid] = H[tid] * d * d;
}

// ---------------------------------------------------------------------------
// Edge scatter: agg[dst] += H[src] * dis[src]*dis[dst]; F/4 threads per edge,
// float4 gathers (L2-resident: 51MB feature matrix < 192MB L2)
// ---------------------------------------------------------------------------
template<int F>
__global__ void k_edge_scatter(const int* __restrict__ src, const int* __restrict__ dst,
                               const float* __restrict__ dis, const float* __restrict__ H,
                               float* __restrict__ agg, long long e_cnt) {
    constexpr int C4 = F / 4;
    long long tid = (long long)blockIdx.x * blockDim.x + threadIdx.x;
    long long e = tid / C4;
    if (e >= e_cnt) return;
    int c4 = (int)(tid - e * C4);
    int s = src[e], d = dst[e];
    float w = dis[s] * dis[d];
    const float4* hp = (const float4*)(H + (size_t)s * F);
    float4 v = hp[c4];
    float* ap = agg + (size_t)d * F + c4 * 4;
    atomicAdd(ap + 0, v.x * w);
    atomicAdd(ap + 1, v.y * w);
    atomicAdd(ap + 2, v.z * w);
    atomicAdd(ap + 3, v.w * w);
}

// ---------------------------------------------------------------------------
// BatchNorm: per-channel sum / sumsq of (agg + bias), then apply + ReLU
// ---------------------------------------------------------------------------
__global__ void k_zero(float* __restrict__ p, int n) {
    int i = blockIdx.x * blockDim.x + threadIdx.x;
    if (i < n) p[i] = 0.0f;
}

__global__ void k_bn_stats(const float* __restrict__ agg, const float* __restrict__ bias,
                           float* __restrict__ stats, int n, int rows_per_blk) {
    int c  = threadIdx.x;                            // 0..127
    int r0 = blockIdx.x * rows_per_blk;
    int r1 = r0 + rows_per_blk; if (r1 > n) r1 = n;
    float b = bias[c];
    float s1 = 0.0f, s2 = 0.0f;
    for (int r = r0; r < r1; ++r) {
        float v = agg[(size_t)r * 128 + c] + b;
        s1 += v; s2 += v * v;
    }
    atomicAdd(&stats[c],       s1);
    atomicAdd(&stats[128 + c], s2);
}

__global__ void k_bn_apply(const float* __restrict__ agg, const float* __restrict__ bias,
                           const float* __restrict__ gamma, const float* __restrict__ beta,
                           const float* __restrict__ stats, float* __restrict__ out, int n) {
    long long tid = (long long)blockIdx.x * blockDim.x + threadIdx.x;
    if (tid >= (long long)n * 128) return;
    int c = (int)(tid & 127);
    float invn = 1.0f / (float)n;
    float mean = stats[c] * invn;
    float var  = stats[128 + c] * invn - mean * mean;
    float rstd = rsqrtf(var + EPS_BN);
    float v = agg[tid] + bias[c];
    float y = (v - mean) * rstd * gamma[c] + beta[c];
    out[tid] = fmaxf(y, 0.0f);
}

// ---------------------------------------------------------------------------
// Wave-per-row softmax over 40 logits (+ bias). wave32 shuffle reductions.
// ---------------------------------------------------------------------------
__global__ void k_softmax40(const float* __restrict__ agg, const float* __restrict__ bias,
                            float* __restrict__ out, int n) {
    int row  = blockIdx.x * (blockDim.x >> 5) + (threadIdx.x >> 5);
    int lane = threadIdx.x & 31;
    if (row >= n) return;
    const float* a = agg + (size_t)row * 40;
    float v0 = a[lane] + bias[lane];
    float v1 = (lane < 8) ? (a[lane + 32] + bias[lane + 32]) : -3.0e38f;
    float m = fmaxf(v0, v1);
    #pragma unroll
    for (int off = 16; off > 0; off >>= 1) m = fmaxf(m, __shfl_xor(m, off, 32));
    float e0 = __expf(v0 - m);
    float e1 = (lane < 8) ? __expf(v1 - m) : 0.0f;
    float s = e0 + e1;
    #pragma unroll
    for (int off = 16; off > 0; off >>= 1) s += __shfl_xor(s, off, 32);
    float inv = 1.0f / s;
    out[(size_t)row * 40 + lane] = e0 * inv;
    if (lane < 8) out[(size_t)row * 40 + lane + 32] = e1 * inv;
}

// ---------------------------------------------------------------------------
// Launch
// ---------------------------------------------------------------------------
static inline unsigned nblk(long long total, int bs) {
    return (unsigned)((total + bs - 1) / bs);
}

extern "C" void kernel_launch(void* const* d_in, const int* in_sizes, int n_in,
                              void* d_out, int out_size, void* d_ws, size_t ws_size,
                              hipStream_t stream) {
    const float* x   = (const float*)d_in[0];
    const int*   ei  = (const int*)d_in[1];
    const float* W1  = (const float*)d_in[2];
    const float* b1  = (const float*)d_in[3];
    const float* g1  = (const float*)d_in[4];
    const float* be1 = (const float*)d_in[5];
    const float* W2  = (const float*)d_in[6];
    const float* b2  = (const float*)d_in[7];
    const float* g2  = (const float*)d_in[8];
    const float* be2 = (const float*)d_in[9];
    const float* W3  = (const float*)d_in[10];
    const float* b3  = (const float*)d_in[11];

    const int       N = in_sizes[0] / 128;
    const long long E = in_sizes[1] / 2;
    const int  HID = 128, OUTC = 40;
    const int* src = ei;            // edge_index[0]
    const int* dst = ei + E;        // edge_index[1]

    // workspace layout
    float*     dis   = (float*)d_ws;                               // N (deg -> dis)
    float*     bufA  = dis + (((size_t)N + 255) & ~(size_t)255);   // N*128
    float*     bufB  = bufA + (size_t)N * HID;                     // N*128
    float*     stats = bufB + (size_t)N * HID;                     // 256
    _Float16*  pw1   = (_Float16*)(stats + 256);                   // 4*8*32*16
    _Float16*  pw2   = pw1 + 4 * 8 * 32 * 16;
    _Float16*  pw3   = pw2 + 4 * 8 * 32 * 16;                      // 4*3*32*16

    float* smOut = (float*)d_out;                                  // [N,40]
    float* hOut  = (float*)d_out + (size_t)N * OUTC;               // [N,128]

    const int rowTiles = N / 16;                                   // N % 16 == 0
    const long long NF  = (long long)N * HID;
    const long long NF3 = (long long)N * OUTC;

    // --- weight fragment packing (once) ------------------------------------
    k_packW<128><<<nblk(4 * 8 * 32 * 16, 256), 256, 0, stream>>>(W1, pw1);
    k_packW<128><<<nblk(4 * 8 * 32 * 16, 256), 256, 0, stream>>>(W2, pw2);
    k_packW<40> <<<nblk(4 * 3 * 32 * 16, 256), 256, 0, stream>>>(W3, pw3);

    // --- normalization -----------------------------------------------------
    k_deg_init <<<nblk(N, 256), 256, 0, stream>>>(dis, N);
    k_deg_edges<<<nblk(E, 256), 256, 0, stream>>>(dst, dis, E);
    k_dis      <<<nblk(N, 256), 256, 0, stream>>>(dis, N);

    // --- layer 1: x -> bufA(H1) -> bufB(agg1) -> BN+ReLU -> bufA -----------
    k_gemm<128>        <<<rowTiles, 256, 0, stream>>>(x, pw1, bufA);
    k_selfloop<128>    <<<nblk(NF, 256), 256, 0, stream>>>(bufA, dis, bufB, N);
    k_edge_scatter<128><<<nblk(E * 32, 256), 256, 0, stream>>>(src, dst, dis, bufA, bufB, E);
    k_zero             <<<1, 256, 0, stream>>>(stats, 256);
    k_bn_stats         <<<200, 128, 0, stream>>>(bufB, b1, stats, N, (N + 199) / 200);
    k_bn_apply         <<<nblk(NF, 256), 256, 0, stream>>>(bufB, b1, g1, be1, stats, bufA, N);

    // --- layer 2: bufA -> bufB(H2) -> bufA(agg2) -> BN+ReLU -> hOut --------
    k_gemm<128>        <<<rowTiles, 256, 0, stream>>>(bufA, pw2, bufB);
    k_selfloop<128>    <<<nblk(NF, 256), 256, 0, stream>>>(bufB, dis, bufA, N);
    k_edge_scatter<128><<<nblk(E * 32, 256), 256, 0, stream>>>(src, dst, dis, bufB, bufA, E);
    k_zero             <<<1, 256, 0, stream>>>(stats, 256);
    k_bn_stats         <<<200, 128, 0, stream>>>(bufA, b2, stats, N, (N + 199) / 200);
    k_bn_apply         <<<nblk(NF, 256), 256, 0, stream>>>(bufA, b2, g2, be2, stats, hOut, N);

    // --- layer 3: hOut -> bufA(H3, N x 40) -> bufB(agg3) -> softmax --------
    k_gemm<40>         <<<rowTiles, 256, 0, stream>>>(hOut, pw3, bufA);
    k_selfloop<40>     <<<nblk(NF3, 256), 256, 0, stream>>>(bufA, dis, bufB, N);
    k_edge_scatter<40> <<<nblk(E * 10, 256), 256, 0, stream>>>(src, dst, dis, bufA, bufB, E);
    k_softmax40        <<<nblk(N, 256), 256, 0, stream>>>(bufB, b3, smOut, N);

    (void)n_in; (void)out_size; (void)ws_size;
}
